// GenerativeTransformer_14834817040610
// MI455X (gfx1250) — compile-verified, matrix operand order checked by reading, not run on hardware
//
#include <hip/hip_runtime.h>

// Transformer forward for MI455X (gfx1250, wave32, WMMA).
// All GEMMs use v_wmma_f32_16x16x32_bf16 (fp32 accumulate), weights converted
// fp32->bf16 on the fly while staging into LDS (compute-bound: AI ~1000 F/B).

#define LNUM 6
#define BB   2
#define SS   1024
#define DD   1024
#define HH   16
#define DKK  64
#define FFD  4096
#define VV   32000

typedef __attribute__((ext_vector_type(16))) __bf16 v16bf;
typedef __attribute__((ext_vector_type(8)))  __bf16 bf16x8;
typedef __attribute__((ext_vector_type(8)))  float  v8f;

__device__ __forceinline__ v8f vzero8() {
  v8f r;
#pragma unroll
  for (int i = 0; i < 8; ++i) r[i] = 0.0f;
  return r;
}

__device__ __forceinline__ v16bf mkfrag(bf16x8 lo, bf16x8 hi) {
  v16bf r;
#pragma unroll
  for (int i = 0; i < 8; ++i) { r[i] = lo[i]; r[i + 8] = hi[i]; }
  return r;
}

// Build a 16-elem bf16 fragment from two runs of 8 consecutive fp32 values.
__device__ __forceinline__ v16bf cvt_frag_f32(const float* p0, const float* p1) {
  float4 a = *(const float4*)p0;
  float4 b = *(const float4*)(p0 + 4);
  float4 c = *(const float4*)p1;
  float4 d = *(const float4*)(p1 + 4);
  v16bf r;
  r[0]  = (__bf16)a.x; r[1]  = (__bf16)a.y; r[2]  = (__bf16)a.z; r[3]  = (__bf16)a.w;
  r[4]  = (__bf16)b.x; r[5]  = (__bf16)b.y; r[6]  = (__bf16)b.z; r[7]  = (__bf16)b.w;
  r[8]  = (__bf16)c.x; r[9]  = (__bf16)c.y; r[10] = (__bf16)c.z; r[11] = (__bf16)c.w;
  r[12] = (__bf16)d.x; r[13] = (__bf16)d.y; r[14] = (__bf16)d.z; r[15] = (__bf16)d.w;
  return r;
}

__device__ __forceinline__ v8f wmma_bf16(v16bf a, v16bf b, v8f c) {
  // D = A(16x32 bf16) x B(32x16 bf16) + C(16x16 f32)
  return __builtin_amdgcn_wmma_f32_16x16x32_bf16(false, a, false, b, (short)0, c,
                                                 false, false);
}

// ---------------------------------------------------------------------------
// Embedding: x[b,s,:] = emb[id]*sqrt(D) + pe[b,:]   (reference's pe[:b] quirk)
// ---------------------------------------------------------------------------
__launch_bounds__(256)
__global__ void embed_kernel(const int* __restrict__ ids, const float* __restrict__ emb,
                             const float* __restrict__ pe, float* __restrict__ x) {
  int row = blockIdx.x;              // b*S + s
  int b   = row / SS;
  int id  = ids[row];
  for (int d = threadIdx.x; d < DD; d += 256)
    x[(size_t)row * DD + d] = emb[(size_t)id * DD + d] * 32.0f + pe[(size_t)b * DD + d];
}

// ---------------------------------------------------------------------------
// GEMM: C[M,N] = act(A[M,K] @ W[K,N] + bias).  fp32 in, bf16 WMMA, fp32 out.
// Block tile 64x128 (8 waves of 32), wave tile 16x64, BK = 32.
// ACT: 0 = none, 1 = exact GELU.
// M,N,K are multiples of 64/128/32 for every call site -> no bounds checks.
// ---------------------------------------------------------------------------
template <int ACT>
__launch_bounds__(256)
__global__ void gemm_bf16_kernel(const float* __restrict__ A, const float* __restrict__ W,
                                 const float* __restrict__ bias, float* __restrict__ C,
                                 int M, int N, int K) {
  __shared__ __attribute__((aligned(16))) __bf16 As[64][40];    // [m][k]
  __shared__ __attribute__((aligned(16))) __bf16 Bs[128][40];   // [n][k] (transposed)

  const int t    = threadIdx.x;
  const int lane = t & 31;
  const int wid  = t >> 5;
  const int half = lane >> 4;
  const int ln   = lane & 15;
  const int m0   = blockIdx.y * 64;
  const int n0   = blockIdx.x * 128;
  const int wm   = (wid >> 1) * 16;   // 4 waves along M
  const int wn   = (wid & 1) * 64;    // 2 waves along N

  v8f acc[4];
#pragma unroll
  for (int n = 0; n < 4; ++n) acc[n] = vzero8();

  for (int k0 = 0; k0 < K; k0 += 32) {
    // Stage A tile (64x32 fp32 -> bf16)
#pragma unroll
    for (int p = 0; p < 2; ++p) {
      int idx = t + p * 256;
      int r   = idx >> 3;
      int c4  = (idx & 7) << 2;
      float4 f = *(const float4*)&A[(size_t)(m0 + r) * K + k0 + c4];
      As[r][c4 + 0] = (__bf16)f.x; As[r][c4 + 1] = (__bf16)f.y;
      As[r][c4 + 2] = (__bf16)f.z; As[r][c4 + 3] = (__bf16)f.w;
    }
    // Stage W tile (32x128 fp32 -> bf16), transposed to [n][k]
#pragma unroll
    for (int p = 0; p < 4; ++p) {
      int idx = t + p * 256;
      int r   = idx >> 5;
      int c4  = (idx & 31) << 2;
      float4 f = *(const float4*)&W[(size_t)(k0 + r) * N + n0 + c4];
      Bs[c4 + 0][r] = (__bf16)f.x; Bs[c4 + 1][r] = (__bf16)f.y;
      Bs[c4 + 2][r] = (__bf16)f.z; Bs[c4 + 3][r] = (__bf16)f.w;
    }
    if (k0 + 32 < K)  // hint next weight tile toward GL2 (global_prefetch_b8)
      __builtin_prefetch(&W[(size_t)(k0 + 32 + (t >> 5)) * N + n0 + ((t & 31) << 2)], 0, 1);
    __syncthreads();

    // A fragment: lane holds row M = ln; K elems {0..7,16..23} / {8..15,24..31} by half
    v16bf afrag;
    {
      const __bf16* row = &As[wm + ln][0];
      bf16x8 lo = *(const bf16x8*)(row + 8 * half);
      bf16x8 hi = *(const bf16x8*)(row + 16 + 8 * half);
      afrag = mkfrag(lo, hi);
    }
#pragma unroll
    for (int n = 0; n < 4; ++n) {
      const __bf16* col = &Bs[wn + n * 16 + ln][0];
      bf16x8 lo = *(const bf16x8*)(col + 8 * half);
      bf16x8 hi = *(const bf16x8*)(col + 16 + 8 * half);
      acc[n] = wmma_bf16(afrag, mkfrag(lo, hi), acc[n]);
    }
    __syncthreads();
  }

  // Epilogue: C layout -> lane ln = N, vgpr j = M (+8 for upper half)
#pragma unroll
  for (int n = 0; n < 4; ++n) {
    int col  = n0 + wn + n * 16 + ln;
    float bv = bias[col];
#pragma unroll
    for (int j = 0; j < 8; ++j) {
      int row = m0 + wm + j + 8 * half;
      float vv = acc[n][j] + bv;
      if (ACT == 1) vv = 0.5f * vv * (1.0f + erff(vv * 0.70710678118654752f));
      C[(size_t)row * N + col] = vv;
    }
  }
}

// ---------------------------------------------------------------------------
// Causal attention, flash style. One wave per 16-row q tile, 4 waves/block.
// scores = q@k^T / 8 (DK=64), online softmax, o += P@V.  WMMA everywhere.
// ---------------------------------------------------------------------------
__launch_bounds__(128)
__global__ void attn_kernel(const float* __restrict__ q, const float* __restrict__ k,
                            const float* __restrict__ v, float* __restrict__ o) {
  __shared__ __attribute__((aligned(16))) __bf16 Plds[4][16][40];

  const int lane  = threadIdx.x & 31;
  const int wid   = threadIdx.x >> 5;
  const int half  = lane >> 4;
  const int ln    = lane & 15;
  const int qtile = blockIdx.x * 4 + wid;
  const int h     = blockIdx.y;
  const int b     = blockIdx.z;
  const int qbase = qtile * 16;
  const size_t base = (size_t)b * SS * DD + (size_t)h * DKK;

  // q tile as two A fragments (16x32 each: d 0..31 and 32..63)
  const float* qrow = q + base + (size_t)(qbase + ln) * DD;
  v16bf qa0 = cvt_frag_f32(qrow + 8 * half,      qrow + 16 + 8 * half);
  v16bf qa1 = cvt_frag_f32(qrow + 32 + 8 * half, qrow + 48 + 8 * half);

  v8f oacc[4];
#pragma unroll
  for (int n = 0; n < 4; ++n) oacc[n] = vzero8();
  float m[8], lsum[8];
#pragma unroll
  for (int j = 0; j < 8; ++j) { m[j] = -1e30f; lsum[j] = 0.0f; }

  for (int kv0 = 0; kv0 <= qbase + 15; kv0 += 32) {
    // k rows as B fragments: lane ln = key column kj
    const float* kr0 = k + base + (size_t)(kv0 + ln) * DD;
    const float* kr1 = kr0 + (size_t)16 * DD;
    v8f s0 = vzero8(), s1 = vzero8();
    s0 = wmma_bf16(qa0, cvt_frag_f32(kr0 + 8 * half,      kr0 + 16 + 8 * half), s0);
    s0 = wmma_bf16(qa1, cvt_frag_f32(kr0 + 32 + 8 * half, kr0 + 48 + 8 * half), s0);
    s1 = wmma_bf16(qa0, cvt_frag_f32(kr1 + 8 * half,      kr1 + 16 + 8 * half), s1);
    s1 = wmma_bf16(qa1, cvt_frag_f32(kr1 + 32 + 8 * half, kr1 + 48 + 8 * half), s1);

    // scale, causal mask, online softmax (row = cross 16 lanes within a half)
#pragma unroll
    for (int j = 0; j < 8; ++j) {
      int qi   = qbase + j + 8 * half;
      float x0 = s0[j] * 0.125f; if (kv0 + ln > qi)      x0 = -1e9f;
      float x1 = s1[j] * 0.125f; if (kv0 + 16 + ln > qi) x1 = -1e9f;
      float t = fmaxf(x0, x1);
      t = fmaxf(t, __shfl_xor(t, 1, 32));
      t = fmaxf(t, __shfl_xor(t, 2, 32));
      t = fmaxf(t, __shfl_xor(t, 4, 32));
      t = fmaxf(t, __shfl_xor(t, 8, 32));
      float mn    = fmaxf(m[j], t);
      float alpha = __expf(m[j] - mn);
      float p0    = __expf(x0 - mn);
      float p1    = __expf(x1 - mn);
      float rs = p0 + p1;
      rs += __shfl_xor(rs, 1, 32);
      rs += __shfl_xor(rs, 2, 32);
      rs += __shfl_xor(rs, 4, 32);
      rs += __shfl_xor(rs, 8, 32);
      lsum[j] = lsum[j] * alpha + rs;
      m[j]    = mn;
#pragma unroll
      for (int n = 0; n < 4; ++n) oacc[n][j] *= alpha;
      // C-layout -> LDS (row-major P tile) for the A-layout reload
      Plds[wid][j + 8 * half][ln]      = (__bf16)p0;
      Plds[wid][j + 8 * half][16 + ln] = (__bf16)p1;
    }
    asm volatile("s_wait_dscnt 0" ::: "memory");  // cross-lane LDS transpose

    // P as A fragment (16x32)
    const __bf16* prow = &Plds[wid][ln][0];
    v16bf pa = mkfrag(*(const bf16x8*)(prow + 8 * half),
                      *(const bf16x8*)(prow + 16 + 8 * half));

    // o += P @ V : V chunk (32 x 64) as 4 B fragments
#pragma unroll
    for (int n = 0; n < 4; ++n) {
      int col = h * DKK + n * 16 + ln;
      v16bf bv;
#pragma unroll
      for (int j = 0; j < 16; ++j) {
        int kk = (j < 8) ? (j + 8 * half) : (j + 8 + 8 * half);
        bv[j] = (__bf16)v[((size_t)b * SS + kv0 + kk) * DD + col];
      }
      oacc[n] = wmma_bf16(pa, bv, oacc[n]);
    }
  }

  // normalize and store o (row = qbase + j + 8*half, col = lane N)
#pragma unroll
  for (int n = 0; n < 4; ++n) {
    int col = h * DKK + n * 16 + ln;
#pragma unroll
    for (int j = 0; j < 8; ++j) {
      int row = qbase + j + 8 * half;
      o[((size_t)b * SS + row) * DD + col] = oacc[n][j] / lsum[j];
    }
  }
}

// ---------------------------------------------------------------------------
// x = LayerNorm(x + r) * gamma + beta   (one block per row, D=1024)
// ---------------------------------------------------------------------------
__launch_bounds__(256)
__global__ void add_ln_kernel(float* __restrict__ x, const float* __restrict__ r,
                              const float* __restrict__ gam, const float* __restrict__ bet) {
  __shared__ float red[256];
  int row = blockIdx.x, t = threadIdx.x;
  float vv[4];
  float s = 0.0f;
#pragma unroll
  for (int i = 0; i < 4; ++i) {
    int d = t + i * 256;
    vv[i] = x[(size_t)row * DD + d] + r[(size_t)row * DD + d];
    s += vv[i];
  }
  red[t] = s; __syncthreads();
  for (int off = 128; off > 0; off >>= 1) {
    if (t < off) red[t] += red[t + off];
    __syncthreads();
  }
  float mu = red[0] * (1.0f / DD);
  __syncthreads();
  s = 0.0f;
#pragma unroll
  for (int i = 0; i < 4; ++i) { vv[i] -= mu; s += vv[i] * vv[i]; }
  red[t] = s; __syncthreads();
  for (int off = 128; off > 0; off >>= 1) {
    if (t < off) red[t] += red[t + off];
    __syncthreads();
  }
  float rs = rsqrtf(red[0] * (1.0f / DD) + 1e-5f);
#pragma unroll
  for (int i = 0; i < 4; ++i) {
    int d = t + i * 256;
    x[(size_t)row * DD + d] = vv[i] * rs * gam[d] + bet[d];
  }
}

// ---------------------------------------------------------------------------
extern "C" void kernel_launch(void* const* d_in, const int* in_sizes, int n_in,
                              void* d_out, int out_size, void* d_ws, size_t ws_size,
                              hipStream_t stream) {
  const int*   ids  = (const int*)  d_in[0];
  const float* emb  = (const float*)d_in[1];
  const float* pe   = (const float*)d_in[2];
  const float* Wq   = (const float*)d_in[3];
  const float* bq   = (const float*)d_in[4];
  const float* Wk   = (const float*)d_in[5];
  const float* bk   = (const float*)d_in[6];
  const float* Wv   = (const float*)d_in[7];
  const float* bv   = (const float*)d_in[8];
  const float* Wo   = (const float*)d_in[9];
  const float* bo   = (const float*)d_in[10];
  const float* ln1s = (const float*)d_in[11];
  const float* ln1b = (const float*)d_in[12];
  const float* W1   = (const float*)d_in[13];
  const float* b1   = (const float*)d_in[14];
  const float* W2   = (const float*)d_in[15];
  const float* b2   = (const float*)d_in[16];
  const float* ln2s = (const float*)d_in[17];
  const float* ln2b = (const float*)d_in[18];
  const float* Wout = (const float*)d_in[19];
  const float* bout = (const float*)d_in[20];

  const int ROWS = BB * SS;  // 2048
  float* ws  = (float*)d_ws; // needs ~80 MB
  float* x   = ws;
  float* tmp = x   + (size_t)ROWS * DD;
  float* qb  = tmp + (size_t)ROWS * DD;
  float* kb  = qb  + (size_t)ROWS * DD;
  float* vb  = kb  + (size_t)ROWS * DD;
  float* ao  = vb  + (size_t)ROWS * DD;
  float* hb  = ao  + (size_t)ROWS * DD;   // ROWS * FFD

  dim3 blk(256);
  dim3 gD(DD / 128, ROWS / 64);     // D-output GEMMs
  dim3 gF(FFD / 128, ROWS / 64);    // FFN up-projection
  dim3 gV(VV / 128, ROWS / 64);     // vocab projection

  embed_kernel<<<dim3(ROWS), blk, 0, stream>>>(ids, emb, pe, x);

  for (int l = 0; l < LNUM; ++l) {
    const float* wq = Wq + (size_t)l * DD * DD;
    const float* wk = Wk + (size_t)l * DD * DD;
    const float* wv = Wv + (size_t)l * DD * DD;
    const float* wo = Wo + (size_t)l * DD * DD;
    const float* w1 = W1 + (size_t)l * DD * FFD;
    const float* w2 = W2 + (size_t)l * FFD * DD;

    gemm_bf16_kernel<0><<<gD, blk, 0, stream>>>(x, wq, bq + (size_t)l * DD, qb, ROWS, DD, DD);
    gemm_bf16_kernel<0><<<gD, blk, 0, stream>>>(x, wk, bk + (size_t)l * DD, kb, ROWS, DD, DD);
    gemm_bf16_kernel<0><<<gD, blk, 0, stream>>>(x, wv, bv + (size_t)l * DD, vb, ROWS, DD, DD);

    attn_kernel<<<dim3(SS / 64, HH, BB), dim3(128), 0, stream>>>(qb, kb, vb, ao);

    gemm_bf16_kernel<0><<<gD, blk, 0, stream>>>(ao, wo, bo + (size_t)l * DD, tmp, ROWS, DD, DD);
    add_ln_kernel<<<dim3(ROWS), blk, 0, stream>>>(x, tmp, ln1s + (size_t)l * DD, ln1b + (size_t)l * DD);

    gemm_bf16_kernel<1><<<gF, blk, 0, stream>>>(x, w1, b1 + (size_t)l * FFD, hb, ROWS, FFD, DD);
    gemm_bf16_kernel<0><<<gD, blk, 0, stream>>>(hb, w2, b2 + (size_t)l * DD, tmp, ROWS, DD, FFD);
    add_ln_kernel<<<dim3(ROWS), blk, 0, stream>>>(x, tmp, ln2s + (size_t)l * DD, ln2b + (size_t)l * DD);
  }

  gemm_bf16_kernel<0><<<gV, blk, 0, stream>>>(x, Wout, bout, (float*)d_out, ROWS, DD, VV);
}
